// APPNP_88476326298056
// MI455X (gfx1250) — compile-verified, matrix operand order checked by reading, not run on hardware
//
#include <hip/hip_runtime.h>

// ---------------------------------------------------------------------------
// APPNP on MI455X (gfx1250, wave32):
//   h = relu(x W1^T + b1); h = relu(h W2^T + b2); h = h W3^T + b3
//   p = h;  10x: p = 0.5*(A_hat p) + 0.5*h,   A_hat = D^-1 (A + I)
// GEMMs: v_wmma_f32_16x16x32_bf16, LDS-staged, b128 fragment loads.
// Grid ordered n-fastest so sibling n-blocks hit the A tile in L2.
// Propagation: float4-vectorized f32 atomic scatter (p/tmp are L2-resident).
// ---------------------------------------------------------------------------

typedef __attribute__((ext_vector_type(16))) __bf16 v16bf;
typedef __attribute__((ext_vector_type(8)))  __bf16 v8bf;
typedef __attribute__((ext_vector_type(8)))  float  v8f;

#define NFEAT  512
#define NHID   256
#define NCLASS 40

// f32 -> bf16 round-to-nearest-even via native hardware conversion
__device__ __forceinline__ __bf16 f2bf(float f) {
  return (__bf16)f;
}

__device__ __forceinline__ v8bf cvt8(const float4 a, const float4 b) {
  v8bf o;
  o[0] = f2bf(a.x); o[1] = f2bf(a.y); o[2] = f2bf(a.z); o[3] = f2bf(a.w);
  o[4] = f2bf(b.x); o[5] = f2bf(b.y); o[6] = f2bf(b.z); o[7] = f2bf(b.w);
  return o;
}

// ---------------------------------------------------------------------------
// Tiled WMMA GEMM:  Out[M x Nout] = act(A[M x K] * W[Nout x K]^T + bias)
// 256 threads = 8 waves; block tile 64x64; LDS K-stage = 64 (2 WMMA K-steps).
// Wave w: m-subtile (w&3), n-subtiles {(w>>2)*2, (w>>2)*2+1}.
// As and Bs both stored [64 rows][72] (row = m for A, n for B): every
// fragment half is one contiguous 16B LDS load (ds_load_b128); 144B row
// stride (36 banks) keeps the 16 lanes bank-conflict-free.
// blockIdx.x = n-tile (fastest) so the same A tile is L2-hot across n.
// ---------------------------------------------------------------------------
template<int K, bool A_F32, bool RELU, bool OUT_BF16>
__global__ __launch_bounds__(256) void gemm_bf16_wmma(
    const void* __restrict__ Aptr, const float* __restrict__ W,
    const float* __restrict__ bias, void* __restrict__ Out,
    int M, int Nout)
{
  constexpr int BM = 64, BN = 64, BK = 64;
  constexpr int LDP = BK + 8;            // 72 bf16 = 144B row stride
  __shared__ __bf16 As[BM][LDP];         // As[m][k]
  __shared__ __bf16 Bs[BN][LDP];         // Bs[n][k] = W[n0+n][k0+k]

  const int n0   = blockIdx.x * BN;      // n fastest-varying
  const int m0   = blockIdx.y * BM;
  const int t    = threadIdx.x;
  const int wave = t >> 5;
  const int lid  = t & 31;
  const int half = lid >> 4;
  const int ml   = lid & 15;
  const int mt   = wave & 3;
  const int nt0  = (wave >> 2) * 2;

  // staging assignment: thread t owns 16 contiguous elements of one row
  const int sr = t >> 2;                 // row 0..63
  const int sc = (t & 3) * 16;           // col 0,16,32,48

  const int ga  = m0 + sr;
  const int gac = (ga < M) ? ga : (M - 1);         // clamped, always valid
  const bool aok = (ga < M);
  const int gb  = n0 + sr;
  const int gbc = (gb < Nout) ? gb : (Nout - 1);
  const bool bok = (gb < Nout);

  v8f acc0 = {}; v8f acc1 = {};

  for (int k0 = 0; k0 < K; k0 += BK) {
    // ---- stage A row-chunk: 16 elements -> two b128 LDS stores ----
    v8bf a0, a1;
    if (A_F32) {
      const float4* ap = (const float4*)((const float*)Aptr + (size_t)gac * K + k0 + sc);
      float4 f0 = ap[0], f1 = ap[1], f2 = ap[2], f3 = ap[3];
      if (!aok) { f0 = f1 = f2 = f3 = make_float4(0.f, 0.f, 0.f, 0.f); }
      a0 = cvt8(f0, f1); a1 = cvt8(f2, f3);
    } else {
      const v8bf* ap = (const v8bf*)((const __bf16*)Aptr + (size_t)gac * K + k0 + sc);
      a0 = ap[0]; a1 = ap[1];
      if (!aok) { a0 = (v8bf)(__bf16)0; a1 = (v8bf)(__bf16)0; }
    }
    *(v8bf*)&As[sr][sc]     = a0;
    *(v8bf*)&As[sr][sc + 8] = a1;

    // ---- stage B row-chunk (weights, f32 -> bf16) ----
    {
      const float4* bp = (const float4*)(W + (size_t)gbc * K + k0 + sc);
      float4 f0 = bp[0], f1 = bp[1], f2 = bp[2], f3 = bp[3];
      if (!bok) { f0 = f1 = f2 = f3 = make_float4(0.f, 0.f, 0.f, 0.f); }
      *(v8bf*)&Bs[sr][sc]     = cvt8(f0, f1);
      *(v8bf*)&Bs[sr][sc + 8] = cvt8(f2, f3);
    }
    __syncthreads();

    const int arow = mt * 16 + ml;
    const int bc0  = nt0 * 16 + ml;
    const int bc1  = bc0 + 16;
    #pragma unroll
    for (int s = 0; s < 2; ++s) {        // two 32-wide WMMA K-steps
      const int kb = s * 32 + half * 8;  // lane's K start; holds kb..kb+7, kb+16..kb+23
      v8bf alo = *(const v8bf*)&As[arow][kb];
      v8bf ahi = *(const v8bf*)&As[arow][kb + 16];
      v8bf b0l = *(const v8bf*)&Bs[bc0][kb];
      v8bf b0h = *(const v8bf*)&Bs[bc0][kb + 16];
      v8bf b1l = *(const v8bf*)&Bs[bc1][kb];
      v8bf b1h = *(const v8bf*)&Bs[bc1][kb + 16];
      v16bf a  = __builtin_shufflevector(alo, ahi, 0,1,2,3,4,5,6,7,8,9,10,11,12,13,14,15);
      v16bf b0 = __builtin_shufflevector(b0l, b0h, 0,1,2,3,4,5,6,7,8,9,10,11,12,13,14,15);
      v16bf b1 = __builtin_shufflevector(b1l, b1h, 0,1,2,3,4,5,6,7,8,9,10,11,12,13,14,15);
      // EXEC all-ones in this region (no divergence) per WMMA rules.
      acc0 = __builtin_amdgcn_wmma_f32_16x16x32_bf16(false, a, false, b0,
                                                     (short)0, acc0, false, false);
      acc1 = __builtin_amdgcn_wmma_f32_16x16x32_bf16(false, a, false, b1,
                                                     (short)0, acc1, false, false);
    }
    __syncthreads();
  }

  // ---- epilogue: C layout lane(half,ml): n = ml, m = half*8 + v ----
  #pragma unroll
  for (int v = 0; v < 8; ++v) {
    int gm = m0 + mt * 16 + half * 8 + v;
    if (gm >= M) continue;
    int gn0 = n0 + nt0 * 16 + ml;
    int gn1 = gn0 + 16;
    if (gn0 < Nout) {
      float c = acc0[v] + bias[gn0];
      if (RELU) c = fmaxf(c, 0.0f);
      if (OUT_BF16) ((__bf16*)Out)[(size_t)gm * Nout + gn0] = f2bf(c);
      else          ((float*) Out)[(size_t)gm * Nout + gn0] = c;
    }
    if (gn1 < Nout) {
      float c = acc1[v] + bias[gn1];
      if (RELU) c = fmaxf(c, 0.0f);
      if (OUT_BF16) ((__bf16*)Out)[(size_t)gm * Nout + gn1] = f2bf(c);
      else          ((float*) Out)[(size_t)gm * Nout + gn1] = c;
    }
  }
}

// ---------------------------------------------------------------------------
// Propagation helpers
// ---------------------------------------------------------------------------
__global__ void fill_kernel(float* __restrict__ p, float val, int n) {
  int i = blockIdx.x * blockDim.x + threadIdx.x;
  if (i < n) p[i] = val;
}

__global__ void fill4_kernel(float4* __restrict__ p, int n4) {
  int i = blockIdx.x * blockDim.x + threadIdx.x;
  if (i < n4) p[i] = make_float4(0.f, 0.f, 0.f, 0.f);
}

__global__ void deg_count_kernel(const int* __restrict__ rows,
                                 float* __restrict__ deg, int E) {
  int e = blockIdx.x * blockDim.x + threadIdx.x;
  if (e < E) atomicAdd(&deg[rows[e]], 1.0f);
}

__global__ void deg_inv_kernel(float* __restrict__ deg, int n) {
  int i = blockIdx.x * blockDim.x + threadIdx.x;
  if (i < n) deg[i] = 1.0f / deg[i];   // deg >= 1 (self loop), never zero
}

// one thread per (edge, 4-feature group): tmp[r,f..f+3] += deg_inv[r]*p[c,f..f+3]
__global__ void scatter_kernel(const int* __restrict__ rows,
                               const int* __restrict__ cols,
                               const float* __restrict__ deg_inv,
                               const float* __restrict__ p,
                               float* __restrict__ tmp, int total) {
  int gid = blockIdx.x * blockDim.x + threadIdx.x;
  if (gid >= total) return;
  int e = gid / (NCLASS / 4);
  int g = gid - e * (NCLASS / 4);
  int r = rows[e];
  int c = cols[e];
  float w = deg_inv[r];
  float4 pv = ((const float4*)(p + (size_t)c * NCLASS))[g];
  float* dst = tmp + (size_t)r * NCLASS + g * 4;
  atomicAdd(dst + 0, w * pv.x);
  atomicAdd(dst + 1, w * pv.y);
  atomicAdd(dst + 2, w * pv.z);
  atomicAdd(dst + 3, w * pv.w);
}

// p' = 0.5*(tmp + deg_inv[i]*p) + 0.5*h   (self-loop folded in analytically)
__global__ void combine_kernel(const float4* __restrict__ tmp,
                               const float4* __restrict__ p,
                               const float4* __restrict__ h,
                               const float* __restrict__ deg_inv,
                               float4* __restrict__ out, int total4) {
  int gid = blockIdx.x * blockDim.x + threadIdx.x;
  if (gid >= total4) return;
  int i = gid / (NCLASS / 4);
  float w = deg_inv[i];
  float4 t = tmp[gid], pv = p[gid], hv = h[gid];
  float4 o;
  o.x = 0.5f * (t.x + w * pv.x) + 0.5f * hv.x;
  o.y = 0.5f * (t.y + w * pv.y) + 0.5f * hv.y;
  o.z = 0.5f * (t.z + w * pv.z) + 0.5f * hv.z;
  o.w = 0.5f * (t.w + w * pv.w) + 0.5f * hv.w;
  out[gid] = o;
}

// ---------------------------------------------------------------------------
extern "C" void kernel_launch(void* const* d_in, const int* in_sizes, int n_in,
                              void* d_out, int out_size, void* d_ws, size_t ws_size,
                              hipStream_t stream) {
  const float* x    = (const float*)d_in[0];
  const int*   edge = (const int*)  d_in[1];
  const float* W1   = (const float*)d_in[2];
  const float* b1   = (const float*)d_in[3];
  const float* W2   = (const float*)d_in[4];
  const float* b2   = (const float*)d_in[5];
  const float* W3   = (const float*)d_in[6];
  const float* b3   = (const float*)d_in[7];

  const int N = in_sizes[0] / NFEAT;
  const int E = in_sizes[1] / 2;
  const int* rows = edge;        // edge_index[0]
  const int* cols = edge + E;    // edge_index[1]

  // workspace carve-up (256B aligned)
  char* ws = (char*)d_ws;
  size_t off = 0;
  auto carve = [&](size_t bytes) -> void* {
    void* p = ws + off;
    off += (bytes + 255) & ~((size_t)255);
    return p;
  };
  float* deg = (float*)carve((size_t)N * sizeof(float));            // -> deg_inv
  void*  h1  =          carve((size_t)N * NHID * 2);                // bf16
  void*  h2  =          carve((size_t)N * NHID * 2);                // bf16
  float* hf  = (float*)carve((size_t)N * NCLASS * sizeof(float));   // MLP output
  float* pA  = (float*)carve((size_t)N * NCLASS * sizeof(float));
  float* pB  = (float*)carve((size_t)N * NCLASS * sizeof(float));
  float* tmp = (float*)carve((size_t)N * NCLASS * sizeof(float));

  const int n40   = N * NCLASS;
  const int n40_4 = n40 / 4;

  // ---- degree (self-loop contributes the initial 1.0) ----
  fill_kernel<<<(N + 255) / 256, 256, 0, stream>>>(deg, 1.0f, N);
  deg_count_kernel<<<(E + 255) / 256, 256, 0, stream>>>(rows, deg, E);
  deg_inv_kernel<<<(N + 255) / 256, 256, 0, stream>>>(deg, N);

  // ---- MLP via WMMA GEMMs (grid: n fastest for A-operand L2 reuse) ----
  dim3 blk(256);
  dim3 g1((NHID   + 63) / 64, (N + 63) / 64);
  dim3 g3((NCLASS + 63) / 64, (N + 63) / 64);
  gemm_bf16_wmma<NFEAT, true,  true,  true ><<<g1, blk, 0, stream>>>(x,  W1, b1, h1, N, NHID);
  gemm_bf16_wmma<NHID,  false, true,  true ><<<g1, blk, 0, stream>>>(h1, W2, b2, h2, N, NHID);
  gemm_bf16_wmma<NHID,  false, false, false><<<g3, blk, 0, stream>>>(h2, W3, b3, hf, N, NCLASS);

  // ---- 10 rounds of p = 0.5*A_hat p + 0.5*h (iter 0 reads h directly) ----
  const float* pin = hf;
  const int scatter_total = E * (NCLASS / 4);
  for (int it = 0; it < 10; ++it) {
    fill4_kernel<<<(n40_4 + 255) / 256, 256, 0, stream>>>((float4*)tmp, n40_4);
    scatter_kernel<<<(scatter_total + 255) / 256, 256, 0, stream>>>(
        rows, cols, deg, pin, tmp, scatter_total);
    float* pout = (it == 9) ? (float*)d_out : ((it & 1) ? pB : pA);
    combine_kernel<<<(n40_4 + 255) / 256, 256, 0, stream>>>(
        (const float4*)tmp, (const float4*)pin, (const float4*)hf, deg,
        (float4*)pout, n40_4);
    pin = pout;
  }
}
